// GatedTransformerXLLayer_63591285784722
// MI455X (gfx1250) — matrix operand order repaired and checked by reading, not compiled, not run on hardware
//
#include <hip/hip_runtime.h>

// ---------------------------------------------------------------------------
// Gated Transformer-XL layer for MI455X (gfx1250, wave32, WMMA bf16->f32).
// D=1024, H=16, HD=64, FF=4096, CUR=256, MEM=256, BS=32, FULL=512.
// ---------------------------------------------------------------------------

typedef __bf16 bf16_t;
typedef __bf16 v16bf __attribute__((ext_vector_type(16)));
typedef __bf16 v8bf  __attribute__((ext_vector_type(8)));
typedef float  v8f   __attribute__((ext_vector_type(8)));

#define D_      1024
#define H_      16
#define HD_     64
#define FF_     4096
#define CUR_    256
#define MEM_    256
#define BS_     32
#define FULL_   512
#define SCALE_  0.125f  // 1/sqrt(64)

// ---------------- fragment helpers ----------------
// A-operand (16x32 bf16): lane r=L&15 holds row r; K = 16*hi + [0..15]
// contiguous (hi = L>>4). Two 16B loads.
__device__ __forceinline__ v16bf load_a_frag(const bf16_t* p) {
  v8bf lo = *reinterpret_cast<const v8bf*>(p);
  v8bf hi = *reinterpret_cast<const v8bf*>(p + 8);
  return __builtin_shufflevector(lo, hi, 0,1,2,3,4,5,6,7,8,9,10,11,12,13,14,15);
}
// B-operand (32x16 bf16): lane n=L&15 holds column n, K = 16*hi + [0..15]
// contiguous. From row-major B^T (N x K) this is one contiguous 32B load.
__device__ __forceinline__ v16bf load_b_frag(const bf16_t* p) {
  return *reinterpret_cast<const v16bf*>(p);
}
__device__ __forceinline__ v8f wmma_bf16(v16bf a, v16bf b, v8f c) {
  return __builtin_amdgcn_wmma_f32_16x16x32_bf16(false, a, false, b, (short)0, c,
                                                 false, false);
}
__device__ __forceinline__ v8f zero8() {
  v8f z;
#pragma unroll
  for (int k = 0; k < 8; ++k) z[k] = 0.0f;
  return z;
}
__device__ __forceinline__ float gelu_f(float x) {
  return 0.5f * x * (1.0f + erff(x * 0.70710678118654752f));
}
__device__ __forceinline__ float sigmoid_f(float x) {
  return 1.0f / (1.0f + expf(-x));
}

// ---------------- generic WMMA GEMM: C(MxN) = A(MxK) @ Bt(NxK)^T ------------
// mode 0: C=acc   1: C+=acc   2: Cb=bf16(acc+bias)   3: C=acc+bias
// mode 4: Cb=bf16(gelu(acc+bias))
__global__ void __launch_bounds__(32)
k_gemm(const bf16_t* __restrict__ A, int lda,
       const bf16_t* __restrict__ Bt, int ldb,
       float* __restrict__ C, bf16_t* __restrict__ Cb, int ldc,
       const float* __restrict__ bias, int K, int mode) {
  const int lane = threadIdx.x;
  const int r16 = lane & 15, hi = lane >> 4;
  const int m0 = blockIdx.x * 32, n0 = blockIdx.y * 32;

  v8f acc[2][2];
#pragma unroll
  for (int mi = 0; mi < 2; ++mi)
#pragma unroll
    for (int nj = 0; nj < 2; ++nj) acc[mi][nj] = zero8();

  const bf16_t* a0 = A + (size_t)(m0 + r16) * lda + hi * 16;
  const bf16_t* a1 = A + (size_t)(m0 + 16 + r16) * lda + hi * 16;
  const bf16_t* b0 = Bt + (size_t)(n0 + r16) * ldb + hi * 16;
  const bf16_t* b1 = Bt + (size_t)(n0 + 16 + r16) * ldb + hi * 16;

  for (int k0 = 0; k0 < K; k0 += 32) {
    v16bf af0 = load_a_frag(a0 + k0);
    v16bf af1 = load_a_frag(a1 + k0);
    v16bf bf0 = load_b_frag(b0 + k0);
    v16bf bf1 = load_b_frag(b1 + k0);
    acc[0][0] = wmma_bf16(af0, bf0, acc[0][0]);
    acc[0][1] = wmma_bf16(af0, bf1, acc[0][1]);
    acc[1][0] = wmma_bf16(af1, bf0, acc[1][0]);
    acc[1][1] = wmma_bf16(af1, bf1, acc[1][1]);
  }

#pragma unroll
  for (int mi = 0; mi < 2; ++mi)
#pragma unroll
    for (int nj = 0; nj < 2; ++nj) {
      const int col = n0 + nj * 16 + r16;
      const float bv = (mode >= 2) ? bias[col] : 0.0f;
#pragma unroll
      for (int v = 0; v < 8; ++v) {
        const int row = m0 + mi * 16 + hi * 8 + v;
        const size_t idx = (size_t)row * ldc + col;
        const float x = acc[mi][nj][v];
        if (mode == 0)      C[idx] = x;
        else if (mode == 1) C[idx] += x;
        else if (mode == 2) Cb[idx] = (bf16_t)(x + bv);
        else if (mode == 3) C[idx] = x + bv;
        else                Cb[idx] = (bf16_t)gelu_f(x + bv);
      }
    }
}

// ---------------- LayerNorm over D=1024 (optional concat of two sources) ----
__global__ void __launch_bounds__(256)
k_layernorm(const float* __restrict__ A, long rowsA, const float* __restrict__ B2,
            const float* __restrict__ g, const float* __restrict__ bta,
            bf16_t* __restrict__ out) {
  const long row = blockIdx.x;
  const float* src = (B2 && row >= rowsA) ? (B2 + (row - rowsA) * (long)D_)
                                          : (A + row * (long)D_);
  __shared__ float sh[256];
  float s = 0.0f, ss = 0.0f;
  for (int c = threadIdx.x; c < D_; c += 256) {
    float x = src[c];
    s += x;
    ss += x * x;
  }
  sh[threadIdx.x] = s;
  __syncthreads();
  for (int o = 128; o > 0; o >>= 1) {
    if (threadIdx.x < (unsigned)o) sh[threadIdx.x] += sh[threadIdx.x + o];
    __syncthreads();
  }
  const float mean = sh[0] * (1.0f / D_);
  __syncthreads();
  sh[threadIdx.x] = ss;
  __syncthreads();
  for (int o = 128; o > 0; o >>= 1) {
    if (threadIdx.x < (unsigned)o) sh[threadIdx.x] += sh[threadIdx.x + o];
    __syncthreads();
  }
  const float var = sh[0] * (1.0f / D_) - mean * mean;
  const float inv = rsqrtf(var + 1e-5f);
  for (int c = threadIdx.x; c < D_; c += 256)
    out[row * (long)D_ + c] = (bf16_t)(((src[c] - mean) * inv) * g[c] + bta[c]);
}

// ---------------- small elementwise / layout kernels ------------------------
__global__ void __launch_bounds__(256)
k_wt(const float* __restrict__ W, bf16_t* __restrict__ Wt, int K, int N) {
  const int k = blockIdx.x * 16 + threadIdx.x;
  const int n = blockIdx.y * 16 + threadIdx.y;
  if (k < K && n < N) Wt[(size_t)n * K + k] = (bf16_t)W[(size_t)k * N + n];
}

__global__ void __launch_bounds__(256)
k_f2b(const float* __restrict__ src, bf16_t* __restrict__ dst, long n) {
  const long i = blockIdx.x * (long)blockDim.x + threadIdx.x;
  if (i < n) dst[i] = (bf16_t)src[i];
}

// q (CUR,BS,H,HD) f32  ->  qu/qv  [b][h][i][d] bf16 with u/v added
__global__ void __launch_bounds__(256)
k_qprep(const float* __restrict__ q, const float* __restrict__ u,
        const float* __restrict__ v, bf16_t* __restrict__ qu,
        bf16_t* __restrict__ qv) {
  const long idx = blockIdx.x * (long)blockDim.x + threadIdx.x;
  if (idx >= (long)CUR_ * BS_ * H_ * HD_) return;
  const int d = idx & 63, h = (idx >> 6) & 15, b = (idx >> 10) & 31;
  const int i = (int)(idx >> 15);
  const float x = q[idx];
  const long o = ((long)(b * H_ + h) * CUR_ + i) * HD_ + d;
  qu[o] = (bf16_t)(x + u[h * HD_ + d]);
  qv[o] = (bf16_t)(x + v[h * HD_ + d]);
}

// val half of kv (FULL,BS,2*H*HD) bf16 -> vt [b][h][d][j] bf16
__global__ void __launch_bounds__(256)
k_vtrans(const bf16_t* __restrict__ kv, bf16_t* __restrict__ vt) {
  const long idx = blockIdx.x * (long)blockDim.x + threadIdx.x;
  if (idx >= (long)FULL_ * BS_ * H_ * HD_) return;
  const int d = idx & 63, h = (idx >> 6) & 15, b = (idx >> 10) & 31;
  const int j = (int)(idx >> 15);
  vt[((long)(b * H_ + h) * HD_ + d) * FULL_ + j] =
      kv[((long)j * BS_ + b) * 2048 + 1024 + h * HD_ + d];
}

__global__ void __launch_bounds__(256)
k_gate_r(const float* __restrict__ tr, const float* __restrict__ x,
         bf16_t* __restrict__ rx, long n) {
  const long i = blockIdx.x * (long)blockDim.x + threadIdx.x;
  if (i >= n) return;
  rx[i] = (bf16_t)(sigmoid_f(tr[i]) * x[i]);
}

__global__ void __launch_bounds__(256)
k_gate_final(const float* __restrict__ tz, const float* __restrict__ th,
             const float* __restrict__ x, const float* __restrict__ bg,
             float* __restrict__ outf, bf16_t* __restrict__ outb, long n) {
  const long i = blockIdx.x * (long)blockDim.x + threadIdx.x;
  if (i >= n) return;
  const float z = sigmoid_f(tz[i] - bg[i & (D_ - 1)]);
  const float hv = tanhf(th[i]);
  const float o = (1.0f - z) * x[i] + z * hv;
  outf[i] = o;
  if (outb) outb[i] = (bf16_t)o;
}

// ---------------- fused flash-attention with rel-shift ----------------------
// One wave per (32-row i-tile, b, h).  rel_shift: pos[i,j] = pos_raw[i, j+255-i]
// for unmasked j<=i+MEM; window columns for a 32x32 (i,j) tile are exactly the
// two 32-aligned column tiles starting at c0 = j0 + 224 - i0.
__global__ void __launch_bounds__(32)
k_attn(const bf16_t* __restrict__ qu, const bf16_t* __restrict__ qv,
       const bf16_t* __restrict__ kvbf, const bf16_t* __restrict__ rproj,
       const bf16_t* __restrict__ vt, bf16_t* __restrict__ a1) {
  const int lane = threadIdx.x;
  const int r16 = lane & 15, hi = lane >> 4;
  const int it = blockIdx.x, b = blockIdx.y, h = blockIdx.z;
  const int i0 = it * 32;

  __shared__ float sS[32][33];
  __shared__ float sP[32][65];
  __shared__ __attribute__((aligned(32))) bf16_t sPb[32][32];
  __shared__ float sScale[32];

  // Preload Q fragments (A operand), rows i0..i0+31, K = HD = 64 (2 k-steps)
  v16bf aQu[2][2], aQv[2][2];
  {
    const bf16_t* quB = qu + ((size_t)(b * H_ + h) * CUR_ + i0) * HD_;
    const bf16_t* qvB = qv + ((size_t)(b * H_ + h) * CUR_ + i0) * HD_;
#pragma unroll
    for (int mi = 0; mi < 2; ++mi)
#pragma unroll
      for (int ks = 0; ks < 2; ++ks) {
        const size_t off = (size_t)(mi * 16 + r16) * HD_ + ks * 32 + hi * 16;
        aQu[mi][ks] = load_a_frag(quB + off);
        aQv[mi][ks] = load_a_frag(qvB + off);
      }
  }

  v8f o[2][4];
#pragma unroll
  for (int mi = 0; mi < 2; ++mi)
#pragma unroll
    for (int nd = 0; nd < 4; ++nd) o[mi][nd] = zero8();

  float m = -__builtin_inff(), l = 0.0f;
  const int i_row = i0 + lane;

  int jmax = i0 + 31 + MEM_;
  if (jmax > FULL_ - 1) jmax = FULL_ - 1;
  const int jtEnd = jmax / 32 + 1;

  for (int jt = 0; jt < jtEnd; ++jt) {
    const int j0 = jt * 32;

    // ---- content tile: (Q+u) @ K^T ----
    {
      v8f c[2][2];
#pragma unroll
      for (int mi = 0; mi < 2; ++mi)
#pragma unroll
        for (int nj = 0; nj < 2; ++nj) c[mi][nj] = zero8();
#pragma unroll
      for (int ks = 0; ks < 2; ++ks) {
        v16bf bk0 = load_b_frag(kvbf + ((size_t)(j0 + r16) * BS_ + b) * 2048 +
                                h * HD_ + ks * 32 + hi * 16);
        v16bf bk1 = load_b_frag(kvbf + ((size_t)(j0 + 16 + r16) * BS_ + b) * 2048 +
                                h * HD_ + ks * 32 + hi * 16);
#pragma unroll
        for (int mi = 0; mi < 2; ++mi) {
          c[mi][0] = wmma_bf16(aQu[mi][ks], bk0, c[mi][0]);
          c[mi][1] = wmma_bf16(aQu[mi][ks], bk1, c[mi][1]);
        }
      }
#pragma unroll
      for (int mi = 0; mi < 2; ++mi)
#pragma unroll
        for (int nj = 0; nj < 2; ++nj)
#pragma unroll
          for (int v = 0; v < 8; ++v)
            sS[mi * 16 + hi * 8 + v][nj * 16 + r16] = c[mi][nj][v];
    }

    // ---- pos window: (Q+v) @ R^T for cols [c0, c0+64) ----
    {
      const int c0 = j0 + 224 - i0;
#pragma unroll
      for (int pt = 0; pt < 2; ++pt) {
        const int p0 = c0 + pt * 32;
        v8f pc[2][2];
#pragma unroll
        for (int mi = 0; mi < 2; ++mi)
#pragma unroll
          for (int nj = 0; nj < 2; ++nj) pc[mi][nj] = zero8();
        if (p0 < FULL_) {
#pragma unroll
          for (int ks = 0; ks < 2; ++ks) {
            v16bf br0 = load_b_frag(rproj + (size_t)(p0 + r16) * (H_ * HD_) +
                                    h * HD_ + ks * 32 + hi * 16);
            v16bf br1 = load_b_frag(rproj + (size_t)(p0 + 16 + r16) * (H_ * HD_) +
                                    h * HD_ + ks * 32 + hi * 16);
#pragma unroll
            for (int mi = 0; mi < 2; ++mi) {
              pc[mi][0] = wmma_bf16(aQv[mi][ks], br0, pc[mi][0]);
              pc[mi][1] = wmma_bf16(aQv[mi][ks], br1, pc[mi][1]);
            }
          }
        }
#pragma unroll
        for (int mi = 0; mi < 2; ++mi)
#pragma unroll
          for (int nj = 0; nj < 2; ++nj)
#pragma unroll
            for (int v = 0; v < 8; ++v)
              sP[mi * 16 + hi * 8 + v][pt * 32 + nj * 16 + r16] = pc[mi][nj][v];
      }
    }
    __syncthreads();

    // ---- online softmax (lane = row) ----
    {
      float vmax = -__builtin_inff();
      float vals[32];
#pragma unroll
      for (int dj = 0; dj < 32; ++dj) {
        const int j = j0 + dj;
        float x = -__builtin_inff();
        if (j <= i_row + MEM_)
          x = (sS[lane][dj] + sP[lane][dj + 31 - lane]) * SCALE_;
        vals[dj] = x;
        vmax = fmaxf(vmax, x);
      }
      const float mnew = fmaxf(m, vmax);
      const float scale = __expf(m - mnew);
      float lsum = 0.0f;
#pragma unroll
      for (int dj = 0; dj < 32; ++dj) {
        const float p = __expf(vals[dj] - mnew);
        sPb[lane][dj] = (bf16_t)p;
        lsum += p;
      }
      l = l * scale + lsum;
      m = mnew;
      sScale[lane] = scale;
    }
    __syncthreads();

    // ---- rescale O and accumulate P @ V ----
    {
#pragma unroll
      for (int mi = 0; mi < 2; ++mi) {
        float sc[8];
#pragma unroll
        for (int v = 0; v < 8; ++v) sc[v] = sScale[mi * 16 + hi * 8 + v];
#pragma unroll
        for (int nd = 0; nd < 4; ++nd)
#pragma unroll
          for (int v = 0; v < 8; ++v) o[mi][nd][v] *= sc[v];
      }
      v16bf aP0 = load_a_frag(&sPb[r16][hi * 16]);
      v16bf aP1 = load_a_frag(&sPb[16 + r16][hi * 16]);
      const bf16_t* vtB = vt + (size_t)(b * H_ + h) * HD_ * FULL_;
#pragma unroll
      for (int nd = 0; nd < 4; ++nd) {
        v16bf bv = load_b_frag(vtB + (size_t)(nd * 16 + r16) * FULL_ + j0 + hi * 16);
        o[0][nd] = wmma_bf16(aP0, bv, o[0][nd]);
        o[1][nd] = wmma_bf16(aP1, bv, o[1][nd]);
      }
    }
    __syncthreads();
  }

  // ---- normalize and store a1[(i*BS+b)*1024 + h*64 + d] ----
  sScale[lane] = 1.0f / l;
  __syncthreads();
#pragma unroll
  for (int mi = 0; mi < 2; ++mi) {
    float sc[8];
#pragma unroll
    for (int v = 0; v < 8; ++v) sc[v] = sScale[mi * 16 + hi * 8 + v];
#pragma unroll
    for (int nd = 0; nd < 4; ++nd)
#pragma unroll
      for (int v = 0; v < 8; ++v) {
        const int row = i0 + mi * 16 + hi * 8 + v;
        const int col = h * HD_ + nd * 16 + r16;
        a1[((size_t)row * BS_ + b) * D_ + col] = (bf16_t)(o[mi][nd][v] * sc[v]);
      }
  }
}

// ---------------------------------------------------------------------------
extern "C" void kernel_launch(void* const* d_in, const int* in_sizes, int n_in,
                              void* d_out, int out_size, void* d_ws, size_t ws_size,
                              hipStream_t stream) {
  (void)in_sizes; (void)n_in; (void)out_size; (void)ws_size;

  const float* f_inputs = (const float*)d_in[0];
  const float* f_posemb = (const float*)d_in[1];
  const float* f_u      = (const float*)d_in[2];
  const float* f_v      = (const float*)d_in[3];
  const float* f_memory = (const float*)d_in[4];
  // d_in[5]: boolean mask (recomputed analytically in-kernel)
  const float* ln1_g = (const float*)d_in[6];
  const float* ln1_b = (const float*)d_in[7];
  const float* ln2_g = (const float*)d_in[8];
  const float* ln2_b = (const float*)d_in[9];
  const float* W_kv = (const float*)d_in[10]; const float* b_kv = (const float*)d_in[11];
  const float* W_q  = (const float*)d_in[12]; const float* b_q  = (const float*)d_in[13];
  const float* W_ps = (const float*)d_in[14]; const float* b_ps = (const float*)d_in[15];
  const float* W_o  = (const float*)d_in[16]; const float* b_o  = (const float*)d_in[17];
  const float* g1w[6]; for (int i = 0; i < 6; ++i) g1w[i] = (const float*)d_in[18 + i];
  const float* g1_bg = (const float*)d_in[24];
  const float* g2w[6]; for (int i = 0; i < 6; ++i) g2w[i] = (const float*)d_in[25 + i];
  const float* g2_bg = (const float*)d_in[31];
  const float* W_m1 = (const float*)d_in[32]; const float* b_m1 = (const float*)d_in[33];
  const float* W_m2 = (const float*)d_in[34]; const float* b_m2 = (const float*)d_in[35];

  const long Ntok = (long)CUR_ * BS_ * D_;        // 8 Mi elements
  const long Nful = (long)FULL_ * BS_ * D_;       // 16 Mi elements

  // ---- workspace bump allocator ----
  size_t off = 0;
  auto alloc = [&](size_t bytes) -> void* {
    void* p = (char*)d_ws + off;
    off += (bytes + 255) & ~(size_t)255;
    return p;
  };
  bf16_t* wt_kv = (bf16_t*)alloc((size_t)2048 * 1024 * 2);
  bf16_t* wt_q  = (bf16_t*)alloc((size_t)1024 * 1024 * 2);
  bf16_t* wt_ps = (bf16_t*)alloc((size_t)1024 * 1024 * 2);
  bf16_t* wt_o  = (bf16_t*)alloc((size_t)1024 * 1024 * 2);
  bf16_t* wt_g1[6]; for (int i = 0; i < 6; ++i) wt_g1[i] = (bf16_t*)alloc((size_t)1024 * 1024 * 2);
  bf16_t* wt_g2[6]; for (int i = 0; i < 6; ++i) wt_g2[i] = (bf16_t*)alloc((size_t)1024 * 1024 * 2);
  bf16_t* wt_m1 = (bf16_t*)alloc((size_t)4096 * 1024 * 2);
  bf16_t* wt_m2 = (bf16_t*)alloc((size_t)4096 * 1024 * 2);
  bf16_t* pos_bf = (bf16_t*)alloc((size_t)FULL_ * D_ * 2);
  bf16_t* xin_bf = (bf16_t*)alloc((size_t)Ntok * 2);
  bf16_t* x1_bf  = (bf16_t*)alloc((size_t)Nful * 2);
  bf16_t* kv_bf  = (bf16_t*)alloc((size_t)Nful * 2 * 2);  // FULL*BS x 2048
  bf16_t* vt_bf  = (bf16_t*)alloc((size_t)Nful * 2);
  float*  q_f32  = (float*)alloc((size_t)Ntok * 4);
  bf16_t* qu_bf  = (bf16_t*)alloc((size_t)Ntok * 2);
  bf16_t* qv_bf  = (bf16_t*)alloc((size_t)Ntok * 2);
  bf16_t* r_bf   = (bf16_t*)alloc((size_t)FULL_ * D_ * 2);
  bf16_t* a1_bf  = (bf16_t*)alloc((size_t)Ntok * 2);
  bf16_t* y1_bf  = (bf16_t*)alloc((size_t)Ntok * 2);
  bf16_t* rx_bf  = (bf16_t*)alloc((size_t)Ntok * 2);
  float*  o1_f32 = (float*)alloc((size_t)Ntok * 4);
  bf16_t* o1_bf  = (bf16_t*)alloc((size_t)Ntok * 2);
  bf16_t* x2_bf  = (bf16_t*)alloc((size_t)Ntok * 2);
  bf16_t* y2_bf  = (bf16_t*)alloc((size_t)Ntok * 2);
  // buffer reuse (lifetimes are disjoint in the launch schedule below)
  float*  t_r   = (float*)x1_bf;   // x1 dead after kv GEMM
  float*  t_z   = (float*)vt_bf;   // vt dead after attention
  float*  t_h   = q_f32;           // q dead after k_qprep
  bf16_t* m1_bf = kv_bf;           // kv dead after attention

  auto gemm = [&](const bf16_t* A, int lda, const bf16_t* Bt, int ldb, float* C,
                  bf16_t* Cb, int ldc, const float* bias, int M, int N, int K,
                  int mode) {
    dim3 g(M / 32, N / 32);
    k_gemm<<<g, 32, 0, stream>>>(A, lda, Bt, ldb, C, Cb, ldc, bias, K, mode);
  };
  auto wt = [&](const float* W, bf16_t* Wt, int K, int N) {
    dim3 g((K + 15) / 16, (N + 15) / 16), blk(16, 16);
    k_wt<<<g, blk, 0, stream>>>(W, Wt, K, N);
  };
  auto ew = [&](long n) { return dim3((unsigned)((n + 255) / 256)); };

  // ---- weight transposition to bf16 (N x K) ----
  wt(W_kv, wt_kv, 1024, 2048);
  wt(W_q,  wt_q,  1024, 1024);
  wt(W_ps, wt_ps, 1024, 1024);
  wt(W_o,  wt_o,  1024, 1024);
  for (int i = 0; i < 6; ++i) wt(g1w[i], wt_g1[i], 1024, 1024);
  for (int i = 0; i < 6; ++i) wt(g2w[i], wt_g2[i], 1024, 1024);
  wt(W_m1, wt_m1, 1024, 4096);
  wt(W_m2, wt_m2, 4096, 1024);

  k_f2b<<<ew((long)FULL_ * D_), 256, 0, stream>>>(f_posemb, pos_bf, (long)FULL_ * D_);
  k_f2b<<<ew(Ntok), 256, 0, stream>>>(f_inputs, xin_bf, Ntok);

  // ---- LN1 over [memory; inputs] ----
  k_layernorm<<<FULL_ * BS_, 256, 0, stream>>>(f_memory, (long)MEM_ * BS_, f_inputs,
                                               ln1_g, ln1_b, x1_bf);

  // ---- projections ----
  gemm(x1_bf, 1024, wt_kv, 1024, nullptr, kv_bf, 2048, b_kv, FULL_ * BS_, 2048, 1024, 2);
  gemm(xin_bf, 1024, wt_q, 1024, q_f32, nullptr, 1024, b_q, CUR_ * BS_, 1024, 1024, 3);
  gemm(pos_bf, 1024, wt_ps, 1024, nullptr, r_bf, 1024, b_ps, FULL_, 1024, 1024, 2);

  k_qprep<<<ew(Ntok), 256, 0, stream>>>(q_f32, f_u, f_v, qu_bf, qv_bf);
  k_vtrans<<<ew(Nful), 256, 0, stream>>>(kv_bf, vt_bf);

  // ---- fused attention ----
  k_attn<<<dim3(CUR_ / 32, BS_, H_), 32, 0, stream>>>(qu_bf, qv_bf, kv_bf, r_bf,
                                                      vt_bf, a1_bf);

  // ---- output projection + GELU -> y1 ----
  gemm(a1_bf, 1024, wt_o, 1024, nullptr, y1_bf, 1024, b_o, CUR_ * BS_, 1024, 1024, 4);

  // ---- GRU gate 1 (x = inputs, y = y1) ----
  gemm(y1_bf, 1024, wt_g1[0], 1024, t_r, nullptr, 1024, nullptr, CUR_ * BS_, 1024, 1024, 0);
  gemm(xin_bf, 1024, wt_g1[1], 1024, t_r, nullptr, 1024, nullptr, CUR_ * BS_, 1024, 1024, 1);
  gemm(y1_bf, 1024, wt_g1[2], 1024, t_z, nullptr, 1024, nullptr, CUR_ * BS_, 1024, 1024, 0);
  gemm(xin_bf, 1024, wt_g1[3], 1024, t_z, nullptr, 1024, nullptr, CUR_ * BS_, 1024, 1024, 1);
  gemm(y1_bf, 1024, wt_g1[4], 1024, t_h, nullptr, 1024, nullptr, CUR_ * BS_, 1024, 1024, 0);
  k_gate_r<<<ew(Ntok), 256, 0, stream>>>(t_r, f_inputs, rx_bf, Ntok);
  gemm(rx_bf, 1024, wt_g1[5], 1024, t_h, nullptr, 1024, nullptr, CUR_ * BS_, 1024, 1024, 1);
  k_gate_final<<<ew(Ntok), 256, 0, stream>>>(t_z, t_h, f_inputs, g1_bg, o1_f32,
                                             o1_bf, Ntok);

  // ---- LN2 + MLP (GELU on both layers) ----
  k_layernorm<<<CUR_ * BS_, 256, 0, stream>>>(o1_f32, (long)CUR_ * BS_, nullptr,
                                              ln2_g, ln2_b, x2_bf);
  gemm(x2_bf, 1024, wt_m1, 1024, nullptr, m1_bf, 4096, b_m1, CUR_ * BS_, 4096, 1024, 4);
  gemm(m1_bf, 4096, wt_m2, 4096, nullptr, y2_bf, 1024, b_m2, CUR_ * BS_, 1024, 4096, 4);

  // ---- GRU gate 2 (x = o1, y = m2) -> d_out ----
  gemm(y2_bf, 1024, wt_g2[0], 1024, t_r, nullptr, 1024, nullptr, CUR_ * BS_, 1024, 1024, 0);
  gemm(o1_bf, 1024, wt_g2[1], 1024, t_r, nullptr, 1024, nullptr, CUR_ * BS_, 1024, 1024, 1);
  gemm(y2_bf, 1024, wt_g2[2], 1024, t_z, nullptr, 1024, nullptr, CUR_ * BS_, 1024, 1024, 0);
  gemm(o1_bf, 1024, wt_g2[3], 1024, t_z, nullptr, 1024, nullptr, CUR_ * BS_, 1024, 1024, 1);
  gemm(y2_bf, 1024, wt_g2[4], 1024, t_h, nullptr, 1024, nullptr, CUR_ * BS_, 1024, 1024, 0);
  k_gate_r<<<ew(Ntok), 256, 0, stream>>>(t_r, o1_f32, rx_bf, Ntok);
  gemm(rx_bf, 1024, wt_g2[5], 1024, t_h, nullptr, 1024, nullptr, CUR_ * BS_, 1024, 1024, 1);
  k_gate_final<<<ew(Ntok), 256, 0, stream>>>(t_z, t_h, o1_f32, g2_bg, (float*)d_out,
                                             nullptr, Ntok);
}
